// AttentionDecoder_40123584479798
// MI455X (gfx1250) — compile-verified, matrix operand order checked by reading, not run on hardware
//
#include <hip/hip_runtime.h>
#include <hip/hip_bf16.h>

typedef __attribute__((ext_vector_type(2)))  float    v2f;
typedef __attribute__((ext_vector_type(8)))  float    v8f;
typedef __attribute__((ext_vector_type(16))) _Float16 v16h;
typedef __attribute__((ext_vector_type(8)))  _Float16 h8v;
typedef __attribute__((ext_vector_type(4)))  unsigned int v4u;
typedef __attribute__((ext_vector_type(4)))  int      v4i;
typedef __attribute__((ext_vector_type(8)))  int      v8i;

#define BV 32000
#define EE 512
#define DD 512
#define HH 512
#define NHH 8
#define HDD 64
#define BB 32
#define TT 64
#define SS 256

__device__ __forceinline__ float sigmoidf_(float x) {
    return 1.0f / (1.0f + __expf(-x));
}

// ---------------------------------------------------------------------------
// fp32 WMMA GEMM:  C[M x N] = A[M x K] * W[N x K]^T + bias[N]  (act: 0=none,1=tanh)
// Wave computes a 32x32 quad of 16x16 tiles; WM x WN waves per WG.
// ---------------------------------------------------------------------------
template <int WM, int WN>
__global__ void wmma_gemm_k(const float* __restrict__ A, int lda,
                            const float* __restrict__ W, int ldw,
                            const float* __restrict__ bias,
                            float* __restrict__ C, int ldc,
                            int K, int act) {
    const int lane = threadIdx.x & 31;
    const int wave = threadIdx.x >> 5;
    const int wm = wave / WN;
    const int wn = wave % WN;
    const int half = lane >> 4;    // 0: K 0..1, 1: K 2..3
    const int l16  = lane & 15;
    const int row0 = blockIdx.x * (WM * 32) + wm * 32;
    const int col0 = blockIdx.y * (WN * 32) + wn * 32;

    v8f acc00 = {0,0,0,0,0,0,0,0};
    v8f acc01 = {0,0,0,0,0,0,0,0};
    v8f acc10 = {0,0,0,0,0,0,0,0};
    v8f acc11 = {0,0,0,0,0,0,0,0};

    const float* Ar0 = A + (size_t)(row0 + l16) * lda + 2 * half;
    const float* Ar1 = A + (size_t)(row0 + 16 + l16) * lda + 2 * half;
    const float* Wr0 = W + (size_t)(col0 + l16) * ldw + 2 * half;
    const float* Wr1 = W + (size_t)(col0 + 16 + l16) * ldw + 2 * half;

    #pragma unroll 2
    for (int k = 0; k < K; k += 4) {
        v2f a0 = *(const v2f*)(Ar0 + k);
        v2f a1 = *(const v2f*)(Ar1 + k);
        v2f b0 = *(const v2f*)(Wr0 + k);
        v2f b1 = *(const v2f*)(Wr1 + k);
        acc00 = __builtin_amdgcn_wmma_f32_16x16x4_f32(false, a0, false, b0, (short)0, acc00, false, false);
        acc01 = __builtin_amdgcn_wmma_f32_16x16x4_f32(false, a0, false, b1, (short)0, acc01, false, false);
        acc10 = __builtin_amdgcn_wmma_f32_16x16x4_f32(false, a1, false, b0, (short)0, acc10, false, false);
        acc11 = __builtin_amdgcn_wmma_f32_16x16x4_f32(false, a1, false, b1, (short)0, acc11, false, false);
    }

    const float bia0 = bias ? bias[col0 + l16] : 0.0f;
    const float bia1 = bias ? bias[col0 + 16 + l16] : 0.0f;

    #pragma unroll
    for (int r = 0; r < 8; ++r) {
        const int m = r + 8 * half;              // C layout: vgpr r, lanes16-31 -> M=r+8
        float v00 = acc00[r] + bia0;
        float v01 = acc01[r] + bia1;
        float v10 = acc10[r] + bia0;
        float v11 = acc11[r] + bia1;
        if (act == 1) { v00 = tanhf(v00); v01 = tanhf(v01); v10 = tanhf(v10); v11 = tanhf(v11); }
        C[(size_t)(row0 + m) * ldc + col0 + l16]           = v00;
        C[(size_t)(row0 + m) * ldc + col0 + 16 + l16]      = v01;
        C[(size_t)(row0 + 16 + m) * ldc + col0 + l16]      = v10;
        C[(size_t)(row0 + 16 + m) * ldc + col0 + 16 + l16] = v11;
    }
}

// ---------------------------------------------------------------------------
// TDM helpers: issue a 2D f16 tile DMA (global -> LDS) via tensor_load_to_lds.
// Tile = tile_rows x 32 halves, row-major; LDS rows padded by 16B (pad_interval
// = 16 DWORDs (code 3), pad_amount = 4 DWORDs (code 3)) -> 80B LDS row stride,
// which makes the 16-lane ds_load_b128 operand fetch bank-conflict-free.
// ---------------------------------------------------------------------------
#define KCHUNK 32
#define LDSROW 40            // halves per LDS row: 32 data + 8 pad

__device__ __forceinline__ unsigned lds_off_(const void* p) {
    // flat shared address: high 32 bits = aperture, low 32 bits = LDS offset
    return (unsigned)(unsigned long long)(uintptr_t)p;
}

__device__ __forceinline__ void tdm_load_tile(unsigned lds_addr, const _Float16* gptr,
                                              int tensor_rows, int K, int tile_rows) {
    unsigned long long ga = (unsigned long long)(uintptr_t)gptr;
    v4u g0;
    g0[0] = 1u;                                              // count=1 (valid user D#)
    g0[1] = lds_addr;                                        // LDS byte address
    g0[2] = (unsigned)(ga & 0xFFFFFFFFull);                  // global_addr[31:0]
    g0[3] = (unsigned)((ga >> 32) & 0x01FFFFFFull) | (2u << 30); // addr[56:32] | type=2
    v8i g1;
    g1[0] = (1 << 16)        // data_size = 1 -> 2 bytes
          | (1 << 20)        // pad_enable
          | (3 << 22)        // pad_interval code 3 = 16 DWORDs (64B of data)
          | (3 << 25);       // pad_amount  code 3 = 4 DWORDs (16B pad)
    g1[1] = (K & 0xFFFF) << 16;                              // tensor_dim0[15:0]
    g1[2] = ((unsigned)K >> 16) | ((tensor_rows & 0xFFFF) << 16);   // dim0 hi | dim1 lo
    g1[3] = (((unsigned)tensor_rows >> 16) & 0xFFFF) | (KCHUNK << 16); // dim1 hi | tile_dim0
    g1[4] = tile_rows & 0xFFFF;                              // tile_dim1 (tile_dim2 = 0)
    g1[5] = K;                                               // tensor_dim0_stride[31:0]
    g1[6] = 0;                                               // stride hi | dim1_stride lo
    g1[7] = 0;
    v4i gz = {0, 0, 0, 0};
#if defined(__clang_major__) && (__clang_major__ >= 23)
    v8i gz8 = {0, 0, 0, 0, 0, 0, 0, 0};
    __builtin_amdgcn_tensor_load_to_lds(g0, g1, gz, gz, gz8, 0);
#else
    __builtin_amdgcn_tensor_load_to_lds(g0, g1, gz, gz, 0);
#endif
}

// ---------------------------------------------------------------------------
// Split-precision f16 WMMA GEMM, TDM-staged + double-buffered:
//   C = (Ahi+Alo)*(Whi+Wlo)^T + bias ~= Ahi*Whi + Ahi*Wlo + Alo*Whi (fp32 acc)
// WG = 64 x 128 output (8 waves, 2x4, each a 32x32 quad). K chunked by 32.
// Wave 0 issues 4 tensor_load_to_lds per chunk; S_WAIT_TENSORCNT + barrier
// pipeline overlaps the DMA of chunk k+1 with WMMA on chunk k.
// ---------------------------------------------------------------------------
__device__ __forceinline__ v16h lds_f16_op(const _Float16* p) {
    v16h r;
    *((h8v*)&r)     = *(const h8v*)(p);
    *(((h8v*)&r)+1) = *(const h8v*)(p + 16);
    return r;
}

__global__ void __launch_bounds__(256)
wmma_gemm_f16x3_tdm_k(const _Float16* __restrict__ Ahi,
                      const _Float16* __restrict__ Alo,
                      const _Float16* __restrict__ Whi,
                      const _Float16* __restrict__ Wlo,
                      const float* __restrict__ bias,
                      float* __restrict__ C, int ldc,
                      int K, int Mtot, int Ntot) {
    __shared__ _Float16 sAh[2][64 * LDSROW];
    __shared__ _Float16 sAl[2][64 * LDSROW];
    __shared__ _Float16 sWh[2][128 * LDSROW];
    __shared__ _Float16 sWl[2][128 * LDSROW];

    const int lane = threadIdx.x & 31;
    const int wave = threadIdx.x >> 5;
    const int wm = wave >> 2;          // 0..1
    const int wn = wave & 3;           // 0..3
    const int half = lane >> 4;
    const int l16  = lane & 15;
    const int row0g = blockIdx.x * 64;
    const int col0g = blockIdx.y * 128;

    v8f acc00 = {0,0,0,0,0,0,0,0};
    v8f acc01 = {0,0,0,0,0,0,0,0};
    v8f acc10 = {0,0,0,0,0,0,0,0};
    v8f acc11 = {0,0,0,0,0,0,0,0};

    const int nk = K / KCHUNK;
    const bool issuer = (threadIdx.x < 32);    // wave 0 owns the TDM

    auto issue_chunk = [&](int kc, int b) {
        const int k0 = kc * KCHUNK;
        tdm_load_tile(lds_off_(&sAh[b][0]), Ahi + (size_t)row0g * K + k0, Mtot, K, 64);
        tdm_load_tile(lds_off_(&sAl[b][0]), Alo + (size_t)row0g * K + k0, Mtot, K, 64);
        tdm_load_tile(lds_off_(&sWh[b][0]), Whi + (size_t)col0g * K + k0, Ntot, K, 128);
        tdm_load_tile(lds_off_(&sWl[b][0]), Wlo + (size_t)col0g * K + k0, Ntot, K, 128);
    };

    auto compute = [&](int b) {
        const int ra0 = (wm * 32 + l16) * LDSROW + 8 * half;
        const int ra1 = ra0 + 16 * LDSROW;
        const int rw0 = (wn * 32 + l16) * LDSROW + 8 * half;
        const int rw1 = rw0 + 16 * LDSROW;
        v16h ah0 = lds_f16_op(&sAh[b][ra0]);
        v16h ah1 = lds_f16_op(&sAh[b][ra1]);
        v16h al0 = lds_f16_op(&sAl[b][ra0]);
        v16h al1 = lds_f16_op(&sAl[b][ra1]);
        v16h bh0 = lds_f16_op(&sWh[b][rw0]);
        v16h bh1 = lds_f16_op(&sWh[b][rw1]);
        v16h bl0 = lds_f16_op(&sWl[b][rw0]);
        v16h bl1 = lds_f16_op(&sWl[b][rw1]);

        acc00 = __builtin_amdgcn_wmma_f32_16x16x32_f16(false, ah0, false, bh0, (short)0, acc00, false, false);
        acc00 = __builtin_amdgcn_wmma_f32_16x16x32_f16(false, ah0, false, bl0, (short)0, acc00, false, false);
        acc00 = __builtin_amdgcn_wmma_f32_16x16x32_f16(false, al0, false, bh0, (short)0, acc00, false, false);

        acc01 = __builtin_amdgcn_wmma_f32_16x16x32_f16(false, ah0, false, bh1, (short)0, acc01, false, false);
        acc01 = __builtin_amdgcn_wmma_f32_16x16x32_f16(false, ah0, false, bl1, (short)0, acc01, false, false);
        acc01 = __builtin_amdgcn_wmma_f32_16x16x32_f16(false, al0, false, bh1, (short)0, acc01, false, false);

        acc10 = __builtin_amdgcn_wmma_f32_16x16x32_f16(false, ah1, false, bh0, (short)0, acc10, false, false);
        acc10 = __builtin_amdgcn_wmma_f32_16x16x32_f16(false, ah1, false, bl0, (short)0, acc10, false, false);
        acc10 = __builtin_amdgcn_wmma_f32_16x16x32_f16(false, al1, false, bh0, (short)0, acc10, false, false);

        acc11 = __builtin_amdgcn_wmma_f32_16x16x32_f16(false, ah1, false, bh1, (short)0, acc11, false, false);
        acc11 = __builtin_amdgcn_wmma_f32_16x16x32_f16(false, ah1, false, bl1, (short)0, acc11, false, false);
        acc11 = __builtin_amdgcn_wmma_f32_16x16x32_f16(false, al1, false, bh1, (short)0, acc11, false, false);
    };

    if (issuer) issue_chunk(0, 0);
    for (int kc = 0; kc + 1 < nk; ++kc) {
        if (issuer) {
            issue_chunk(kc + 1, (kc + 1) & 1);
            __builtin_amdgcn_s_wait_tensorcnt(4);   // chunk kc landed, kc+1 in flight
        }
        __syncthreads();
        compute(kc & 1);
        __syncthreads();    // everyone done reading before chunk kc+2 overwrites
    }
    if (issuer) __builtin_amdgcn_s_wait_tensorcnt(0);
    __syncthreads();
    compute((nk - 1) & 1);

    const int row0 = row0g + wm * 32;
    const int col0 = col0g + wn * 32;
    const float bia0 = bias ? bias[col0 + l16] : 0.0f;
    const float bia1 = bias ? bias[col0 + 16 + l16] : 0.0f;

    #pragma unroll
    for (int r = 0; r < 8; ++r) {
        const int m = r + 8 * half;
        C[(size_t)(row0 + m) * ldc + col0 + l16]           = acc00[r] + bia0;
        C[(size_t)(row0 + m) * ldc + col0 + 16 + l16]      = acc01[r] + bia1;
        C[(size_t)(row0 + 16 + m) * ldc + col0 + l16]      = acc10[r] + bia0;
        C[(size_t)(row0 + 16 + m) * ldc + col0 + 16 + l16] = acc11[r] + bia1;
    }
}

// fp32 -> (f16 hi, f16 lo residual)
__global__ void split16_k(const float* __restrict__ src,
                          _Float16* __restrict__ hi, _Float16* __restrict__ lo, int n) {
    int i = blockIdx.x * blockDim.x + threadIdx.x;
    if (i >= n) return;
    float x = src[i];
    _Float16 h = (_Float16)x;
    hi[i] = h;
    lo[i] = (_Float16)(x - (float)h);
}

// ---------------------------------------------------------------------------
// Per-step attention: one WG per (head, batch). q proj + scores + softmax + ctx.
// ---------------------------------------------------------------------------
__global__ void attn_k(const float* __restrict__ h1,    // (B, 512)
                       const float* __restrict__ wq, const float* __restrict__ bq,
                       const float* __restrict__ kbuf,  // (B, S, 512)
                       const float* __restrict__ vbuf,  // (B, S, 512)
                       float* __restrict__ ctx) {       // (B, 512)
    __shared__ float sh_h[DD];
    __shared__ float sh_q[HDD];
    __shared__ float sh_s[SS];
    __shared__ float sh_r[SS];

    const int tid = threadIdx.x;       // 256 threads
    const int hd  = blockIdx.x;        // head 0..7
    const int b   = blockIdx.y;        // batch 0..31

    sh_h[tid]       = h1[b * DD + tid];
    sh_h[tid + 256] = h1[b * DD + tid + 256];
    __syncthreads();

    if (tid < HDD) {
        const int orow = hd * HDD + tid;
        const float* wr = wq + (size_t)orow * DD;
        float acc = bq[orow];
        for (int kk = 0; kk < DD; ++kk) acc += sh_h[kk] * wr[kk];
        sh_q[tid] = acc;
    }
    __syncthreads();

    {   // scores: thread = source position
        const float* kr = kbuf + ((size_t)(b * SS + tid)) * DD + hd * HDD;
        float acc = 0.0f;
        for (int d = 0; d < HDD; ++d) acc += sh_q[d] * kr[d];
        sh_s[tid] = acc * 0.125f;      // 1/sqrt(64)
    }
    __syncthreads();

    sh_r[tid] = sh_s[tid];
    __syncthreads();
    for (int off = 128; off > 0; off >>= 1) {
        if (tid < off) sh_r[tid] = fmaxf(sh_r[tid], sh_r[tid + off]);
        __syncthreads();
    }
    const float mx = sh_r[0];
    __syncthreads();

    const float e = __expf(sh_s[tid] - mx);
    sh_s[tid] = e;
    sh_r[tid] = e;
    __syncthreads();
    for (int off = 128; off > 0; off >>= 1) {
        if (tid < off) sh_r[tid] += sh_r[tid + off];
        __syncthreads();
    }
    const float inv = 1.0f / sh_r[0];
    __syncthreads();

    {   // ctx: 64 dims x 4 partial groups of 64 source positions
        const int d = tid & 63;
        const int grp = tid >> 6;
        const float* vb = vbuf + ((size_t)(b * SS + grp * 64)) * DD + hd * HDD + d;
        float p = 0.0f;
        for (int s = 0; s < 64; ++s) p += sh_s[grp * 64 + s] * vb[(size_t)s * DD];
        sh_r[tid] = p;
    }
    __syncthreads();
    if (tid < HDD) {
        float c = (sh_r[tid] + sh_r[tid + 64] + sh_r[tid + 128] + sh_r[tid + 192]) * inv;
        ctx[b * DD + hd * HDD + tid] = c;
    }
}

// ---------------------------------------------------------------------------
// Elementwise helpers
// ---------------------------------------------------------------------------
__global__ void emb_gather_k(const int* __restrict__ tgt, const float* __restrict__ emb,
                             float* __restrict__ outp, int total) {
    int idx = blockIdx.x * blockDim.x + threadIdx.x;
    if (idx >= total) return;
    int row = idx / EE, col = idx - row * EE;
    outp[idx] = emb[(size_t)tgt[row] * EE + col];
}

__global__ void pack2_k(const float* __restrict__ wa, int ca,
                        const float* __restrict__ wb, int cb,
                        float* __restrict__ dst, int total) {
    int idx = blockIdx.x * blockDim.x + threadIdx.x;
    if (idx >= total) return;
    int ld = ca + cb;
    int row = idx / ld, col = idx - row * ld;
    dst[idx] = (col < ca) ? wa[(size_t)row * ca + col] : wb[(size_t)row * cb + (col - ca)];
}

__global__ void addvec_k(const float* a, const float* b, float* d, int n) {
    int i = blockIdx.x * blockDim.x + threadIdx.x;
    if (i < n) d[i] = a[i] + b[i];
}

__global__ void zero_k(float* p, int n) {
    int i = blockIdx.x * blockDim.x + threadIdx.x;
    if (i < n) p[i] = 0.0f;
}

__global__ void copy_strided_k(const float* __restrict__ src, int lds,
                               float* __restrict__ dst, int ldd,
                               int rows, int cols) {
    int idx = blockIdx.x * blockDim.x + threadIdx.x;
    if (idx >= rows * cols) return;
    int r = idx / cols, c = idx - r * cols;
    dst[(size_t)r * ldd + c] = src[(size_t)r * lds + c];
}

__global__ void lstm_ew_k(const float* __restrict__ g,   // (32, 2048) gates i,f,g,o
                          float* __restrict__ c,         // (32, 512) in/out
                          float* __restrict__ h,         // (32, 512) out
                          float* __restrict__ h2, int ldh2) {
    int idx = blockIdx.x * blockDim.x + threadIdx.x;
    if (idx >= BB * HH) return;
    int b = idx >> 9, j = idx & 511;
    const float* gr = g + (size_t)b * (4 * HH);
    float ig = sigmoidf_(gr[j]);
    float fg = sigmoidf_(gr[HH + j]);
    float gg = tanhf(gr[2 * HH + j]);
    float og = sigmoidf_(gr[3 * HH + j]);
    float cn = fg * c[idx] + ig * gg;
    c[idx] = cn;
    float hn = og * tanhf(cn);
    h[idx] = hn;
    if (h2) h2[(size_t)b * ldh2 + j] = hn;
}

// ---------------------------------------------------------------------------
extern "C" void kernel_launch(void* const* d_in, const int* in_sizes, int n_in,
                              void* d_out, int out_size, void* d_ws, size_t ws_size,
                              hipStream_t stream) {
    const int*   tgt   = (const int*)d_in[0];
    const float* enc_o = (const float*)d_in[1];
    const float* enc_h = (const float*)d_in[2];
    const float* emb   = (const float*)d_in[3];
    const float* w_q = (const float*)d_in[4],  *b_q = (const float*)d_in[5];
    const float* w_k = (const float*)d_in[6],  *b_k = (const float*)d_in[7];
    const float* w_v = (const float*)d_in[8],  *b_v = (const float*)d_in[9];
    const float* w_o = (const float*)d_in[10], *b_o = (const float*)d_in[11];
    const float* w_ih0 = (const float*)d_in[12], *w_hh0 = (const float*)d_in[13];
    const float* b_ih0 = (const float*)d_in[14], *b_hh0 = (const float*)d_in[15];
    const float* w_ih1 = (const float*)d_in[16], *w_hh1 = (const float*)d_in[17];
    const float* b_ih1 = (const float*)d_in[18], *b_hh1 = (const float*)d_in[19];
    const float* w_hp = (const float*)d_in[20], *b_hp = (const float*)d_in[21];
    const float* w_fc = (const float*)d_in[22], *b_fc = (const float*)d_in[23];
    float* outv = (float*)d_out;
    float* ws   = (float*)d_ws;

    // workspace layout (float units; f16 buffers consume count/2 floats)
    size_t off = 0;
    float* embedded = ws + off; off += (size_t)BB * TT * EE;       // (B,T,E)
    float* kbuf     = ws + off; off += (size_t)BB * SS * DD;       // (B,S,D)
    float* vbuf     = ws + off; off += (size_t)BB * SS * DD;
    float* hcat     = ws + off; off += 2 * BB * HH;                // h0 | h1
    float* c0b      = ws + off; off += BB * HH;
    float* c1b      = ws + off; off += BB * HH;
    float* ctx      = ws + off; off += BB * DD;
    float* X0       = ws + off; off += BB * 1536;                  // [emb | attn_out | h0]
    float* X1       = ws + off; off += BB * 1024;                  // [h0n | h1_prev]
    float* g0       = ws + off; off += BB * 4 * HH;
    float* g1       = ws + off; off += BB * 4 * HH;
    float* W0p      = ws + off; off += (size_t)(4 * HH) * 1536;    // [w_ih0 | w_hh0]
    float* W1p      = ws + off; off += (size_t)(4 * HH) * 1024;    // [w_ih1 | w_hh1]
    float* bs0      = ws + off; off += 4 * HH;
    float* bs1      = ws + off; off += 4 * HH;
    float* outs     = ws + off; off += (size_t)BB * TT * HH;       // (B,T,H)
    const size_t nOuts = (size_t)BB * TT * HH;                     // 1,048,576
    const size_t nWfc  = (size_t)BV * HH;                          // 16,384,000
    _Float16* outs_h = (_Float16*)(ws + off); off += nOuts / 2;
    _Float16* outs_l = (_Float16*)(ws + off); off += nOuts / 2;
    _Float16* wfc_h  = (_Float16*)(ws + off); off += nWfc / 2;
    _Float16* wfc_l  = (_Float16*)(ws + off); off += nWfc / 2;
    float* h0b = hcat;
    float* h1b = hcat + BB * HH;

    const int THR = 256;
    // --- setup ---
    emb_gather_k<<<(BB * TT * EE + THR - 1) / THR, THR, 0, stream>>>(tgt, emb, embedded, BB * TT * EE);
    pack2_k<<<((4 * HH) * 1536 + THR - 1) / THR, THR, 0, stream>>>(w_ih0, EE + DD, w_hh0, HH, W0p, (4 * HH) * 1536);
    pack2_k<<<((4 * HH) * 1024 + THR - 1) / THR, THR, 0, stream>>>(w_ih1, HH, w_hh1, HH, W1p, (4 * HH) * 1024);
    addvec_k<<<(4 * HH + THR - 1) / THR, THR, 0, stream>>>(b_ih0, b_hh0, bs0, 4 * HH);
    addvec_k<<<(4 * HH + THR - 1) / THR, THR, 0, stream>>>(b_ih1, b_hh1, bs1, 4 * HH);
    zero_k<<<(BB * HH + THR - 1) / THR, THR, 0, stream>>>(c0b, BB * HH);
    zero_k<<<(BB * HH + THR - 1) / THR, THR, 0, stream>>>(c1b, BB * HH);
    // split w_fc into f16 hi/lo early (independent of decode loop)
    split16_k<<<(int)((nWfc + THR - 1) / THR), THR, 0, stream>>>(w_fc, wfc_h, wfc_l, (int)nWfc);

    // h0 = tanh(enc_hidden @ w_hp^T + b_hp) : (64 x 512), K=512
    wmma_gemm_k<2, 4><<<dim3(1, DD / 128), THR, 0, stream>>>(enc_h, DD, w_hp, DD, b_hp, hcat, HH, DD, 1);
    // K/V projections: (8192 x 512), K=512 (fp32 WMMA, exact)
    wmma_gemm_k<2, 4><<<dim3(BB * SS / 64, DD / 128), THR, 0, stream>>>(enc_o, DD, w_k, DD, b_k, kbuf, DD, DD, 0);
    wmma_gemm_k<2, 4><<<dim3(BB * SS / 64, DD / 128), THR, 0, stream>>>(enc_o, DD, w_v, DD, b_v, vbuf, DD, DD, 0);

    // --- sequential decode (fp32 WMMA: recurrence must stay exact) ---
    for (int t = 0; t < TT; ++t) {
        attn_k<<<dim3(NHH, BB), THR, 0, stream>>>(h1b, w_q, b_q, kbuf, vbuf, ctx);
        // attn_out = ctx @ w_o^T + b_o  -> X0[:, 512:1024]
        wmma_gemm_k<1, 8><<<dim3(1, DD / 256), THR, 0, stream>>>(ctx, DD, w_o, DD, b_o, X0 + 512, 1536, DD, 0);
        // X0[:, 0:512] = embedded[:, t, :] ; X0[:, 1024:1536] = h0_prev
        copy_strided_k<<<(BB * EE + THR - 1) / THR, THR, 0, stream>>>(embedded + (size_t)t * EE, TT * EE, X0, 1536, BB, EE);
        copy_strided_k<<<(BB * HH + THR - 1) / THR, THR, 0, stream>>>(h0b, HH, X0 + 1024, 1536, BB, HH);
        // gates0 = X0 @ W0p^T + bs0 : (32 x 2048), K=1536
        wmma_gemm_k<1, 8><<<dim3(1, (4 * HH) / 256), THR, 0, stream>>>(X0, 1536, W0p, 1536, bs0, g0, 4 * HH, 1536, 0);
        lstm_ew_k<<<(BB * HH + THR - 1) / THR, THR, 0, stream>>>(g0, c0b, h0b, nullptr, 0);
        // X1 = [h0n | h1_prev]
        copy_strided_k<<<(BB * HH + THR - 1) / THR, THR, 0, stream>>>(h0b, HH, X1, 1024, BB, HH);
        copy_strided_k<<<(BB * HH + THR - 1) / THR, THR, 0, stream>>>(h1b, HH, X1 + 512, 1024, BB, HH);
        // gates1 = X1 @ W1p^T + bs1 : (32 x 2048), K=1024
        wmma_gemm_k<1, 8><<<dim3(1, (4 * HH) / 256), THR, 0, stream>>>(X1, 1024, W1p, 1024, bs1, g1, 4 * HH, 1024, 0);
        // h1 update; also write outs[:, t, :]
        lstm_ew_k<<<(BB * HH + THR - 1) / THR, THR, 0, stream>>>(g1, c1b, h1b, outs + (size_t)t * HH, TT * HH);
    }

    // logits = outs @ w_fc^T + b_fc : (2048 x 32000), K=512 -- TDM-staged f16x3 WMMA
    split16_k<<<(int)((nOuts + THR - 1) / THR), THR, 0, stream>>>(outs, outs_h, outs_l, (int)nOuts);
    wmma_gemm_f16x3_tdm_k<<<dim3(BB * TT / 64, BV / 128), THR, 0, stream>>>(
        outs_h, outs_l, wfc_h, wfc_l, b_fc, outv, BV, HH, BB * TT, BV);
}
